// GCNModel_51762945852140
// MI455X (gfx1250) — matrix operand order repaired and checked by reading, not compiled, not run on hardware
//
#include <hip/hip_runtime.h>
#include <cstdint>

// GCN: out = relu( (D^-1/2 (A+I) D^-1/2) x @ W + b ) @ W2 + b2
// (aggregate-then-transform: scatter in F_IN=128 space, 2x cheaper than H=256)
// N=50000, E=800000, F_IN=128, H=256, C=64

#define N_NODES 50000
#define N_EDGES 800000
#define F_INDIM 128
#define HDIM    256
#define CDIM    64

typedef __attribute__((ext_vector_type(16))) _Float16 v16h;
typedef __attribute__((ext_vector_type(8)))  _Float16 v8h;
typedef __attribute__((ext_vector_type(4)))  _Float16 v4h;
typedef __attribute__((ext_vector_type(8)))  float    v8f;

// ---------------- degree / norm ----------------

__global__ __launch_bounds__(256) void k_init_deg(float* __restrict__ deg) {
    int i = blockIdx.x * 256 + threadIdx.x;
    if (i < N_NODES) deg[i] = 1.0f;   // self-loop contributes 1 to every node
}

__global__ __launch_bounds__(256) void k_degree(const long long* __restrict__ ei,
                                                float* __restrict__ deg) {
    int e = blockIdx.x * 256 + threadIdx.x;   // grid sized exactly to E
    int d = (int)ei[N_EDGES + e];             // dst row of edge_index
    atomicAdd(&deg[d], 1.0f);
}

__global__ __launch_bounds__(256) void k_dinv(float* __restrict__ deg) {
    int i = blockIdx.x * 256 + threadIdx.x;
    if (i < N_NODES) {
        float d = deg[i];
        deg[i] = (d > 0.0f) ? rsqrtf(d) : 0.0f;   // in-place -> dinv
    }
}

// ---------------- seed: xh = f16(x); aggx = dinv[row]^2 * x ----------------
// One streaming pass; fuses the f16 copy for edge gathers with the self-loop
// message (which also initializes aggx, so no memset).

__global__ __launch_bounds__(256) void k_seed(const float* __restrict__ x,
                                              const float* __restrict__ dinv,
                                              _Float16* __restrict__ xh,
                                              float* __restrict__ aggx) {
    int i = (blockIdx.x * 256 + threadIdx.x) * 8;   // grid = 3125 -> exactly N*F_IN
    int row = i >> 7;                               // /F_INDIM
    float di = dinv[row];
    float s  = di * di;
    float4 v0 = *(const float4*)(x + i);
    float4 v1 = *(const float4*)(x + i + 4);
    v8h o;
    o[0] = (_Float16)v0.x; o[1] = (_Float16)v0.y;
    o[2] = (_Float16)v0.z; o[3] = (_Float16)v0.w;
    o[4] = (_Float16)v1.x; o[5] = (_Float16)v1.y;
    o[6] = (_Float16)v1.z; o[7] = (_Float16)v1.w;
    *(v8h*)(xh + i) = o;
    float4 s0 = make_float4(s * v0.x, s * v0.y, s * v0.z, s * v0.w);
    float4 s1 = make_float4(s * v1.x, s * v1.y, s * v1.z, s * v1.w);
    *(float4*)(aggx + i)     = s0;
    *(float4*)(aggx + i + 4) = s1;
}

// ---------------- prep: pack weight into WMMA B-fragment layout (f16) -----
// B-fragment order used here (contiguous k, same as A loads): lane L holds col N = L&15, half j -> K =
// (L>>4)*16 + j.  Packed as dst[((tn*KSTEPS + ks)*32 + lane)*16 + j] so the
// GEMM B-load is one contiguous 32B read per lane.

__global__ __launch_bounds__(256) void k_pack_w(const float* __restrict__ src,
                                                _Float16* __restrict__ dst,
                                                int ksteps, int ldcol) {
    int idx  = blockIdx.x * 256 + threadIdx.x;   // grid sized exactly
    int j    = idx & 15;
    int lane = (idx >> 4) & 31;
    int ks   = (idx >> 9) % ksteps;
    int tn   = (idx >> 9) / ksteps;
    int K    = ks * 32 + (lane >> 4) * 16 + j;
    int col  = tn * 16 + (lane & 15);
    dst[idx] = (_Float16)src[K * ldcol + col];
}

// ---------------- edge aggregation: aggx[dst] += norm * xh[src] ----------------
// One wave per edge in F_IN=128 space: lane moves 4 f16 (one 8B load) + 4 f32
// L2 atomics.  xh (12.8MB) and aggx (25.6MB) are L2-resident on MI455X (192MB
// L2), so this L2-atomic-bound scatter never touches HBM.  102.4M atomics total.

__global__ __launch_bounds__(256) void k_edges(const long long* __restrict__ ei,
                                               const float* __restrict__ dinv,
                                               const _Float16* __restrict__ xh,
                                               float* __restrict__ aggx) {
    const int e    = blockIdx.x * 8 + (threadIdx.x >> 5);  // grid = 100000 -> exactly E
    const int lane = threadIdx.x & 31;
    const int s = (int)ei[e];                    // wave-uniform -> scalar loads
    const int d = (int)ei[N_EDGES + e];
    const float norm = dinv[s] * dinv[d];

    const v4h hv = *(const v4h*)(xh + s * F_INDIM + lane * 4); // coalesced 256B/wave
    float* dst = aggx + d * F_INDIM + lane * 4;
#pragma unroll
    for (int j = 0; j < 4; ++j)
        atomicAdd(dst + j, norm * (float)hv[j]);
}

// ---------------- GEMM1: hf = f16(relu(aggx @ W + b)) ----------------
// One wave per 16x64 output strip (4 N-tiles, 4 accumulators): each A fragment
// feeds 4 WMMAs.  12500 waves = 3125 M-tiles * 4 strips; grid exact.

__global__ __launch_bounds__(128) void k_gemm1(const float* __restrict__ aggx,
                                               const _Float16* __restrict__ Wp,
                                               const float* __restrict__ bias,
                                               _Float16* __restrict__ hf) {
    const int wid     = blockIdx.x * 4 + (threadIdx.x >> 5);  // 0..12499
    const int lane    = threadIdx.x & 31;
    const int tm      = wid >> 2;                // 0..3124
    const int g       = wid & 3;                 // strip of 4 N-tiles
    const int halfsel = lane >> 4;
    const int lrow    = lane & 15;
    const int arow    = tm * 16 + lrow;

    v8f c0 = {}, c1 = {}, c2 = {}, c3 = {};
#pragma unroll
    for (int ks = 0; ks < F_INDIM / 32; ++ks) {
        // A fragment (f32 source): K = ks*32 + halfsel*16 + j
        float4 av[4];
        const float* ap = aggx + arow * F_INDIM + ks * 32 + halfsel * 16;
        av[0] = *(const float4*)(ap);       av[1] = *(const float4*)(ap + 4);
        av[2] = *(const float4*)(ap + 8);   av[3] = *(const float4*)(ap + 12);
        const float* af = reinterpret_cast<const float*>(av);
        v16h a;
#pragma unroll
        for (int j = 0; j < 16; ++j) a[j] = (_Float16)af[j];

#pragma unroll
        for (int t = 0; t < 4; ++t) {
            const int tn = g * 4 + t;
            v16h b = *(const v16h*)(Wp + (((tn * (F_INDIM / 32) + ks) * 32 + lane) << 4));
            v8f& c = (t == 0) ? c0 : (t == 1) ? c1 : (t == 2) ? c2 : c3;
            c = __builtin_amdgcn_wmma_f32_16x16x32_f16(false, a, false, b,
                                                       (short)0, c, false, false);
        }
    }

    // fused bias + ReLU + f32->f16 store into hf (A operand of GEMM2)
#pragma unroll
    for (int t = 0; t < 4; ++t) {
        const v8f& c = (t == 0) ? c0 : (t == 1) ? c1 : (t == 2) ? c2 : c3;
        const int n  = (g * 4 + t) * 16 + lrow;
        const float bn = bias[n];
#pragma unroll
        for (int r = 0; r < 8; ++r) {
            const int m = tm * 16 + r + 8 * halfsel;
            float v = c[r] + bn;
            v = v > 0.0f ? v : 0.0f;
            hf[m * HDIM + n] = (_Float16)v;
        }
    }
}

// ---------------- GEMM2: out = hf @ W2 + b2 ----------------
// One wave per 16x64 output row-strip (all of C=64).  Pure f16 A loads.

__global__ __launch_bounds__(128) void k_gemm2(const _Float16* __restrict__ hf,
                                               const _Float16* __restrict__ W2p,
                                               const float* __restrict__ b2,
                                               float* __restrict__ out) {
    const int wid = blockIdx.x * 4 + (threadIdx.x >> 5);
    if (wid >= N_NODES / 16) return;             // wave-uniform guard (EXEC stays ~0)
    const int lane    = threadIdx.x & 31;
    const int tm      = wid;
    const int halfsel = lane >> 4;
    const int lrow    = lane & 15;
    const int arow    = tm * 16 + lrow;

    v8f c0 = {}, c1 = {}, c2 = {}, c3 = {};
#pragma unroll
    for (int ks = 0; ks < HDIM / 32; ++ks) {
        const _Float16* ap = hf + arow * HDIM + ks * 32 + halfsel * 16;
        v8h a0 = *(const v8h*)(ap);
        v8h a1 = *(const v8h*)(ap + 8);
        v16h a = __builtin_shufflevector(a0, a1, 0, 1, 2, 3, 4, 5, 6, 7,
                                                 8, 9, 10, 11, 12, 13, 14, 15);
#pragma unroll
        for (int t = 0; t < 4; ++t) {
            v16h b = *(const v16h*)(W2p + (((t * (HDIM / 32) + ks) * 32 + lane) << 4));
            v8f& c = (t == 0) ? c0 : (t == 1) ? c1 : (t == 2) ? c2 : c3;
            c = __builtin_amdgcn_wmma_f32_16x16x32_f16(false, a, false, b,
                                                       (short)0, c, false, false);
        }
    }

#pragma unroll
    for (int t = 0; t < 4; ++t) {
        const v8f& c = (t == 0) ? c0 : (t == 1) ? c1 : (t == 2) ? c2 : c3;
        const int n  = t * 16 + lrow;
        const float bn = b2[n];
#pragma unroll
        for (int r = 0; r < 8; ++r) {
            const int m = tm * 16 + r + 8 * halfsel;
            out[m * CDIM + n] = c[r] + bn;
        }
    }
}

// ---------------- launcher ----------------

extern "C" void kernel_launch(void* const* d_in, const int* in_sizes, int n_in,
                              void* d_out, int out_size, void* d_ws, size_t ws_size,
                              hipStream_t stream) {
    (void)in_sizes; (void)n_in; (void)out_size; (void)ws_size;

    const float*     x  = (const float*)d_in[0];
    const long long* ei = (const long long*)d_in[1];   // int64 per reference
    const float*     W  = (const float*)d_in[2];
    const float*     b  = (const float*)d_in[3];
    const float*     W2 = (const float*)d_in[4];
    const float*     b2 = (const float*)d_in[5];
    float*           out = (float*)d_out;

    // workspace layout (256B-aligned offsets):
    //   [0          +200,000)     deg -> dinv (f32)
    //   [200,704    +12,800,000)  xh     (f16, N x F_IN)   edge-gather operand
    //   [13,000,704 +25,600,000)  aggx   (f32, N x F_IN)   scatter target
    //   [38,600,704 +25,600,000)  hf     (f16, N x H)      hidden activations
    //   [64,200,704 +65,536)      Wpack  (f16, 16*4*32*16)
    //   [64,266,240 +32,768)      W2pack (f16, 4*8*32*16)
    char* ws = (char*)d_ws;
    float*    deg    = (float*)ws;
    _Float16* xh     = (_Float16*)(ws + 200704);
    float*    aggx   = (float*)(ws + 13000704);
    _Float16* hf     = (_Float16*)(ws + 38600704);
    _Float16* Wpack  = (_Float16*)(ws + 64200704);
    _Float16* W2pack = (_Float16*)(ws + 64266240);

    // weight pre-swizzle into B-fragment layout
    k_pack_w<<<(16 * 4 * 32 * 16) / 256, 256, 0, stream>>>(W,  Wpack,  F_INDIM / 32, HDIM);
    k_pack_w<<<(4 * 8 * 32 * 16) / 256,  256, 0, stream>>>(W2, W2pack, HDIM / 32,   CDIM);

    // normalization
    k_init_deg<<<(N_NODES + 255) / 256, 256, 0, stream>>>(deg);
    k_degree  <<<N_EDGES / 256,        256, 0, stream>>>(ei, deg);
    k_dinv    <<<(N_NODES + 255) / 256, 256, 0, stream>>>(deg);

    // seed self-loop + f16 copy, scatter, then the two WMMA GEMMs
    k_seed <<<(N_NODES * F_INDIM / 8) / 256, 256, 0, stream>>>(x, deg, xh, aggx);
    k_edges<<<N_EDGES / 8,             256, 0, stream>>>(ei, deg, xh, aggx);
    k_gemm1<<<(N_NODES / 16), 128, 0, stream>>>(aggx, Wpack, b, hf);
    k_gemm2<<<(N_NODES / 16 + 3) / 4,  128, 0, stream>>>(hf, W2pack, b2, out);
}
